// SepConvNet_48593259987285
// MI455X (gfx1250) — matrix operand order repaired and checked by reading, not compiled, not run on hardware
//
#include <hip/hip_runtime.h>
#include <cstdint>
#include <cstddef>

// ---------------------------------------------------------------------------
// Types for CDNA5 WMMA fragments (wave32).
// ---------------------------------------------------------------------------
typedef __attribute__((ext_vector_type(16))) _Float16 v16h;
typedef __attribute__((ext_vector_type(8)))  _Float16 v8h;
typedef __attribute__((ext_vector_type(8)))  float    v8f;

#define TPB 256

static inline int pad32(int c) { return (c + 31) & ~31; }

// ---------------------------------------------------------------------------
// Repack conv weights: fp32 OIHW [Cout][Cin][3][3] -> f16 [9][CoutP][CinP]
// (tap-major, row = output channel, contiguous input channels). Padded
// rows/cols are zeroed so padded channels are algebraically inert.
// ---------------------------------------------------------------------------
__global__ void k_pack_w(const float* __restrict__ w, _Float16* __restrict__ wp,
                         int Cout, int Cin, int CoutP, int CinP, int total)
{
    int i = blockIdx.x * blockDim.x + threadIdx.x;
    if (i >= total) return;
    int ci  = i % CinP;
    int t   = i / CinP;
    int co  = t % CoutP;
    int tap = t / CoutP;
    _Float16 v = (_Float16)0.f;
    if (co < Cout && ci < Cin) {
        int kh = tap / 3, kw = tap % 3;
        v = (_Float16)w[((size_t)(co * Cin + ci) * 3 + kh) * 3 + kw];
    }
    wp[i] = v;
}

// zero-fill f16 buffer (the "zero page" sourced by out-of-bounds halo lanes)
__global__ void k_zero_f16(_Float16* __restrict__ p, int n)
{
    int i = blockIdx.x * blockDim.x + threadIdx.x;
    if (i < n) p[i] = (_Float16)0.f;
}

// fp32 NCHW -> f16 NHWC with channel padding (pad channels = 0)
__global__ void k_nchw_to_nhwc(const float* __restrict__ in, _Float16* __restrict__ out,
                               int B, int C, int HW, int CP, int total)
{
    int i = blockIdx.x * blockDim.x + threadIdx.x;
    if (i >= total) return;
    int c = i % CP;
    int t = i / CP;
    int p = t % HW;
    int b = t / HW;
    _Float16 v = (_Float16)0.f;
    if (c < C) v = (_Float16)in[((size_t)b * C + c) * HW + p];
    out[i] = v;
}

// ---------------------------------------------------------------------------
// Implicit-GEMM 3x3 conv (stride 1, pad 1) on WMMA f16 -> f32.
//   D[co, pix] = sum_{tap, ci} W[tap][co][ci] * In[pix + tap_offset][ci]
//
// Block = 128 threads = 4 waves; block tile = 32 cout x 64 pixels.
// Each wave owns a 16 cout x 32 pixel macro-tile: two 16x16 accumulators
// sharing one A (weight) fragment -> 2 WMMA per 6 b128 loads, no control
// flow in the K loop (halo lanes read from a zero page via cndmask'd
// base pointers hoisted to the tap level).
//
// Fragment layouts used by this kernel (wave32, contiguous K order, not the ISA table order):
//   A (16x32 f16): lane l -> row m = l&15;  elems 0..7  = K half*16+0..7,
//                                           elems 8..15 = K half*16+8..15
//   B (32x16 f16): lane l -> col n = l&15;  elems 0..15 = K half*16+0..15
//   C/D (16x16 f32): lane l -> col n = l&15; vgpr r -> row m = half*8 + r
// NHWC keeps K (channels) contiguous -> aligned b128 loads, no transpose.
// ---------------------------------------------------------------------------
__global__ void __launch_bounds__(128)
k_conv3x3_wmma(const _Float16* __restrict__ in,    // [B][H][W][CinP]
               const _Float16* __restrict__ wpk,   // [9][CoutP][CinP]
               const float*    __restrict__ bias,  // [Cout]
               const _Float16* __restrict__ zpage, // >= CinP+32 zero halves
               _Float16*       __restrict__ out,   // [B][H][W][CoutP]
               int H, int W, int CinP, int Cout, int CoutP, int relu)
{
    const int lane   = threadIdx.x & 31;
    const int wave   = threadIdx.x >> 5;        // 0..3
    const int co_sub = (wave >> 1) * 16;
    const int px_sub = (wave & 1) * 32;
    const int half   = lane >> 4;               // 0: lanes 0-15, 1: lanes 16-31
    const int m      = lane & 15;               // row (A) / col (B,D)

    const int b  = blockIdx.z;
    const int HW = H * W;
    const int p0 = blockIdx.x * 64 + px_sub + m;   // linear output pixels
    const int p1 = p0 + 16;
    const int co_base = blockIdx.y * 32 + co_sub;

    const _Float16* inB = in + (size_t)b * HW * CinP;

    const int y0 = p0 / W, x0 = p0 % W;
    const int y1 = p1 / W, x1 = p1 % W;

    v8f acc0 = {};
    v8f acc1 = {};

    for (int tap = 0; tap < 9; ++tap) {
        const int dy = tap / 3 - 1;
        const int dx = tap % 3 - 1;
        const int py0 = y0 + dy, px0 = x0 + dx;
        const int py1 = y1 + dy, px1 = x1 + dx;
        const bool v0 = (py0 >= 0) & (py0 < H) & (px0 >= 0) & (px0 < W);
        const bool v1 = (py1 >= 0) & (py1 < H) & (px1 >= 0) & (px1 < W);

        const _Float16* aptr = wpk + ((size_t)tap * CoutP + co_base + m) * CinP + half * 16;
        const _Float16* bp0  = (v0 ? inB + (size_t)(py0 * W + px0) * CinP : zpage) + half * 16;
        const _Float16* bp1  = (v1 ? inB + (size_t)(py1 * W + px1) * CinP : zpage) + half * 16;

        for (int ci0 = 0; ci0 < CinP; ci0 += 32) {
            union { v16h v; v8h h[2]; } A, B0, B1;
            A.h[0]  = *(const v8h*)(aptr + ci0);        // K = half*16 + 0..7
            A.h[1]  = *(const v8h*)(aptr + ci0 + 8);    // K = half*16 + 8..15
            B0.h[0] = *(const v8h*)(bp0 + ci0);         // K = half*16 + 0..7
            B0.h[1] = *(const v8h*)(bp0 + ci0 + 8);     // K = half*16 + 8..15
            B1.h[0] = *(const v8h*)(bp1 + ci0);
            B1.h[1] = *(const v8h*)(bp1 + ci0 + 8);
            // D = A x B + C (f32 accumulate); A fragment reused by both tiles.
            acc0 = __builtin_amdgcn_wmma_f32_16x16x32_f16(
                false, A.v, false, B0.v, (short)0, acc0, false, false);
            acc1 = __builtin_amdgcn_wmma_f32_16x16x32_f16(
                false, A.v, false, B1.v, (short)0, acc1, false, false);
        }
    }

    // bias + optional ReLU, convert to f16, one b128 store of 8 channels/tile.
    const int cb = co_base + half * 8;
    union { v8h v; _Float16 e[8]; } o0, o1;
#pragma unroll
    for (int r = 0; r < 8; ++r) {
        int cc = cb + r;
        float bv = (cc < Cout) ? bias[cc] : 0.f;
        float f0 = acc0[r] + bv;
        float f1 = acc1[r] + bv;
        if (relu) { f0 = fmaxf(f0, 0.f); f1 = fmaxf(f1, 0.f); }
        o0.e[r] = (_Float16)f0;
        o1.e[r] = (_Float16)f1;
    }
    *(v8h*)(out + ((size_t)b * HW + p0) * CoutP + cb) = o0.v;
    *(v8h*)(out + ((size_t)b * HW + p1) * CoutP + cb) = o1.v;
}

// ---------------------------------------------------------------------------
// 2x2 average pool, NHWC f16
// ---------------------------------------------------------------------------
__global__ void k_avgpool2(const _Float16* __restrict__ in, _Float16* __restrict__ out,
                           int B, int H, int W, int CP, int total)
{
    int i = blockIdx.x * blockDim.x + threadIdx.x;
    if (i >= total) return;
    int c = i % CP;
    int t = i / CP;
    int Wo = W >> 1, Ho = H >> 1;
    int xo = t % Wo; t /= Wo;
    int yo = t % Ho;
    int b  = t / Ho;
    const _Float16* p = in + (((size_t)b * H + 2 * yo) * W + 2 * xo) * CP + c;
    float s = (float)p[0] + (float)p[CP] + (float)p[(size_t)W * CP] + (float)p[(size_t)W * CP + CP];
    out[i] = (_Float16)(0.25f * s);
}

// ---------------------------------------------------------------------------
// Bilinear x2 upsample, align_corners=True, NHWC f16
// ---------------------------------------------------------------------------
__global__ void k_up2(const _Float16* __restrict__ in, _Float16* __restrict__ out,
                      int B, int H, int W, int CP, int total)
{
    int i = blockIdx.x * blockDim.x + threadIdx.x;
    if (i >= total) return;
    int c = i % CP;
    int t = i / CP;
    int ox = t % (2 * W); t /= (2 * W);
    int oy = t % (2 * H);
    int b  = t / (2 * H);
    float fy = (float)oy * ((float)(H - 1) / (float)(2 * H - 1));
    float fx = (float)ox * ((float)(W - 1) / (float)(2 * W - 1));
    int y0 = (int)fy, x0 = (int)fx;
    int y1 = y0 + 1 < H ? y0 + 1 : H - 1;
    int x1 = x0 + 1 < W ? x0 + 1 : W - 1;
    float ty = fy - (float)y0, tx = fx - (float)x0;
    const _Float16* base = in + (size_t)b * H * W * CP + c;
    float g00 = (float)base[((size_t)y0 * W + x0) * CP];
    float g01 = (float)base[((size_t)y0 * W + x1) * CP];
    float g10 = (float)base[((size_t)y1 * W + x0) * CP];
    float g11 = (float)base[((size_t)y1 * W + x1) * CP];
    float v = (1.f - ty) * ((1.f - tx) * g00 + tx * g01) + ty * ((1.f - tx) * g10 + tx * g11);
    out[i] = (_Float16)v;
}

__global__ void k_add(const _Float16* __restrict__ a, const _Float16* __restrict__ b,
                      _Float16* __restrict__ o, int n)
{
    int i = blockIdx.x * blockDim.x + threadIdx.x;
    if (i < n) o[i] = (_Float16)((float)a[i] + (float)b[i]);
}

// ---------------------------------------------------------------------------
// Separable-kernel apply with replication pad (K=51):
//   out[b,c,i,j] (+)= sum_u V[b,u,i,j] * sum_v img[clamp(i+u-25), clamp(j+v-25)] * H[b,v,i,j]
// V/H stored NHWC f16 with CP=64 (51 valid). fp32 accumulate, fp32 NCHW out.
// ---------------------------------------------------------------------------
__global__ void k_sepconv(const float* __restrict__ frames,  // [B][F][3][H][W]
                          const _Float16* __restrict__ V, const _Float16* __restrict__ Hk,
                          float* __restrict__ out,           // [B][3][H][W]
                          int B, int F, int frame, int H, int W, int accumulate, int total)
{
    int i = blockIdx.x * blockDim.x + threadIdx.x;
    if (i >= total) return;
    int j = i % W; int t = i / W;
    int y = t % H; t /= H;
    int c = t % 3;
    int b = t / 3;
    const float* img = frames + (((size_t)b * F + frame) * 3 + c) * (size_t)H * W;
    const _Float16* vp = V  + ((size_t)(b * H + y) * W + j) * 64;
    const _Float16* hp = Hk + ((size_t)(b * H + y) * W + j) * 64;
    float sum = 0.f;
    for (int u = 0; u < 51; ++u) {
        int iu = y + u - 25;
        iu = iu < 0 ? 0 : (iu > H - 1 ? H - 1 : iu);
        const float* row = img + (size_t)iu * W;
        float hs = 0.f;
        for (int v = 0; v < 51; ++v) {
            int jv = j + v - 25;
            jv = jv < 0 ? 0 : (jv > W - 1 ? W - 1 : jv);
            hs += row[jv] * (float)hp[v];
        }
        sum += hs * (float)vp[u];
    }
    size_t o = ((size_t)(b * 3 + c) * H + y) * W + j;
    if (accumulate) out[o] += sum; else out[o] = sum;
}

// ---------------------------------------------------------------------------
// Host-side orchestration
// ---------------------------------------------------------------------------
namespace {

struct Bump {
    char*  base;
    size_t off;
    void* alloc(size_t bytes) {
        bytes = (bytes + 255) & ~(size_t)255;
        void* r = base + off;
        off += bytes;
        return r;
    }
};

struct Conv {
    const float* bias;
    _Float16*    wp;
    int Cin, Cout, CinP, CoutP;
};

static const _Float16* g_zpage = nullptr;  // set once per kernel_launch call

inline void run_conv(hipStream_t s, const _Float16* in, _Float16* out, const Conv& c,
                     int B, int H, int W, int relu)
{
    dim3 g((unsigned)((H * W) / 64), (unsigned)(c.CoutP / 32), (unsigned)B);
    k_conv3x3_wmma<<<g, 128, 0, s>>>(in, c.wp, c.bias, g_zpage, out,
                                     H, W, c.CinP, c.Cout, c.CoutP, relu);
}

// Basic(): 3x [conv3x3 -> ReLU].  in -> tA -> tB -> out (out may alias tA).
inline void basic3(hipStream_t s, const _Float16* in, _Float16* tA, _Float16* tB,
                   _Float16* out, const Conv* c, int B, int H, int W)
{
    run_conv(s, in, tA, c[0], B, H, W, 1);
    run_conv(s, tA, tB, c[1], B, H, W, 1);
    run_conv(s, tB, out, c[2], B, H, W, 1);
}

// Subnet(): conv-ReLU x3, bilinear x2, conv (no ReLU).  Output at 2H x 2W, CP=64.
inline void subnet(hipStream_t s, const _Float16* in, _Float16* tA, _Float16* tB,
                   _Float16* out, const Conv* c, int B, int H, int W)
{
    run_conv(s, in, tA, c[0], B, H, W, 1);
    run_conv(s, tA, tB, c[1], B, H, W, 1);
    run_conv(s, tB, tA, c[2], B, H, W, 1);            // 64 -> 51 (CP 64)
    int tot = B * 4 * H * W * 64;
    k_up2<<<(tot + TPB - 1) / TPB, TPB, 0, s>>>(tA, tB, B, H, W, 64, tot);
    run_conv(s, tB, out, c[3], B, 2 * H, 2 * W, 0);   // 51 -> 51, no ReLU
}

} // namespace

extern "C" void kernel_launch(void* const* d_in, const int* in_sizes, int n_in,
                              void* d_out, int out_size, void* d_ws, size_t ws_size,
                              hipStream_t stream)
{
    (void)in_sizes; (void)n_in; (void)out_size; (void)ws_size;

    const int B = 2, F = 2, H0 = 256, W0 = 256;
    const size_t HW0 = (size_t)H0 * W0;

    const float* frames = (const float*)d_in[0];

    Bump ws{(char*)d_ws, 0};

    // --- zero page for halo reads (>= CinP_max + 32 halves) ---------------
    {
        const int ZN = 2048;
        _Float16* zp = (_Float16*)ws.alloc((size_t)ZN * sizeof(_Float16));
        k_zero_f16<<<(ZN + TPB - 1) / TPB, TPB, 0, stream>>>(zp, ZN);
        g_zpage = zp;
    }

    // --- weights: inputs in setup_inputs() insertion order, (w,b) pairs ----
    int k = 1;
    auto mk = [&](int cin, int cout) -> Conv {
        Conv c;
        c.Cin = cin; c.Cout = cout;
        c.CinP = pad32(cin); c.CoutP = pad32(cout);
        const float* w = (const float*)d_in[k++];
        c.bias = (const float*)d_in[k++];
        int total = 9 * c.CoutP * c.CinP;
        c.wp = (_Float16*)ws.alloc((size_t)total * sizeof(_Float16));
        k_pack_w<<<(total + TPB - 1) / TPB, TPB, 0, stream>>>(
            w, c.wp, c.Cout, c.Cin, c.CoutP, c.CinP, total);
        return c;
    };

    Conv conv1[3] = { mk(6, 32),    mk(32, 32),   mk(32, 32) };
    Conv conv2[3] = { mk(32, 64),   mk(64, 64),   mk(64, 64) };
    Conv conv3[3] = { mk(64, 128),  mk(128, 128), mk(128, 128) };
    Conv conv4[3] = { mk(128, 256), mk(256, 256), mk(256, 256) };
    Conv conv5[3] = { mk(256, 512), mk(512, 512), mk(512, 512) };
    Conv deconv5[3] = { mk(512, 512), mk(512, 512), mk(512, 512) };
    Conv up5 = mk(512, 512);
    Conv deconv4[3] = { mk(512, 256), mk(256, 256), mk(256, 256) };
    Conv up4 = mk(256, 256);
    Conv deconv3[3] = { mk(256, 128), mk(128, 128), mk(128, 128) };
    Conv up3 = mk(128, 128);
    Conv deconv2[3] = { mk(128, 64), mk(64, 64), mk(64, 64) };
    Conv up2w = mk(64, 64);
    Conv v1[4] = { mk(64, 64), mk(64, 64), mk(64, 51), mk(51, 51) };
    Conv v2[4] = { mk(64, 64), mk(64, 64), mk(64, 51), mk(51, 51) };
    Conv h1[4] = { mk(64, 64), mk(64, 64), mk(64, 51), mk(51, 51) };
    Conv h2[4] = { mk(64, 64), mk(64, 64), mk(64, 51), mk(51, 51) };

    // --- activation buffers (f16 NHWC, padded channels) -------------------
    _Float16* x    = (_Float16*)ws.alloc((size_t)B * HW0 * 32 * 2);        // 256^2 x 32
    _Float16* c1b  = (_Float16*)ws.alloc((size_t)B * HW0 * 32 * 2);        // skip 1
    _Float16* c2b  = (_Float16*)ws.alloc((size_t)B * (HW0 / 4) * 64 * 2);  // skip 2
    _Float16* c3b  = (_Float16*)ws.alloc((size_t)B * (HW0 / 16) * 128 * 2);
    _Float16* c4b  = (_Float16*)ws.alloc((size_t)B * (HW0 / 64) * 256 * 2);
    _Float16* c5b  = (_Float16*)ws.alloc((size_t)B * (HW0 / 256) * 512 * 2);
    _Float16* comb = (_Float16*)ws.alloc((size_t)B * (HW0 / 4) * 64 * 2);  // max comb
    _Float16* tA   = (_Float16*)ws.alloc((size_t)B * HW0 * 64 * 2);
    _Float16* tB   = (_Float16*)ws.alloc((size_t)B * HW0 * 64 * 2);
    _Float16* tC   = (_Float16*)ws.alloc((size_t)B * HW0 * 64 * 2);
    _Float16* Vb   = (_Float16*)ws.alloc((size_t)B * HW0 * 64 * 2);
    _Float16* Hb   = (_Float16*)ws.alloc((size_t)B * HW0 * 64 * 2);

    auto launch1d = [&](int tot) { return dim3((unsigned)((tot + TPB - 1) / TPB)); };

    // --- input: frames [2,2,3,256,256] -> x f16 NHWC [B][H][W][32] (C=6) --
    {
        int tot = B * (int)HW0 * 32;
        k_nchw_to_nhwc<<<launch1d(tot), TPB, 0, stream>>>(frames, x, B, 6, (int)HW0, 32, tot);
    }

    // --- encoder ----------------------------------------------------------
    basic3(stream, x, tA, tB, c1b, conv1, B, 256, 256);
    { int tot = B * 128 * 128 * 32;
      k_avgpool2<<<launch1d(tot), TPB, 0, stream>>>(c1b, tC, B, 256, 256, 32, tot); }
    basic3(stream, tC, tA, tB, c2b, conv2, B, 128, 128);
    { int tot = B * 64 * 64 * 64;
      k_avgpool2<<<launch1d(tot), TPB, 0, stream>>>(c2b, tC, B, 128, 128, 64, tot); }
    basic3(stream, tC, tA, tB, c3b, conv3, B, 64, 64);
    { int tot = B * 32 * 32 * 128;
      k_avgpool2<<<launch1d(tot), TPB, 0, stream>>>(c3b, tC, B, 64, 64, 128, tot); }
    basic3(stream, tC, tA, tB, c4b, conv4, B, 32, 32);
    { int tot = B * 16 * 16 * 256;
      k_avgpool2<<<launch1d(tot), TPB, 0, stream>>>(c4b, tC, B, 32, 32, 256, tot); }
    basic3(stream, tC, tA, tB, c5b, conv5, B, 16, 16);
    { int tot = B * 8 * 8 * 512;
      k_avgpool2<<<launch1d(tot), TPB, 0, stream>>>(c5b, tC, B, 16, 16, 512, tot); }

    // --- decoder: d = basic(p); comb = relu(conv(up2(d))) + skip ----------
    // level 5 (8x8 -> 16x16, 512ch)
    basic3(stream, tC, tA, tB, tA, deconv5, B, 8, 8);
    { int tot = B * 16 * 16 * 512;
      k_up2<<<launch1d(tot), TPB, 0, stream>>>(tA, tB, B, 8, 8, 512, tot);
      run_conv(stream, tB, tC, up5, B, 16, 16, 1);
      k_add<<<launch1d(tot), TPB, 0, stream>>>(tC, c5b, comb, tot); }
    // level 4 (16x16 -> 32x32, 512->256)
    basic3(stream, comb, tA, tB, tA, deconv4, B, 16, 16);
    { int tot = B * 32 * 32 * 256;
      k_up2<<<launch1d(tot), TPB, 0, stream>>>(tA, tB, B, 16, 16, 256, tot);
      run_conv(stream, tB, tC, up4, B, 32, 32, 1);
      k_add<<<launch1d(tot), TPB, 0, stream>>>(tC, c4b, comb, tot); }
    // level 3 (32x32 -> 64x64, 256->128)
    basic3(stream, comb, tA, tB, tA, deconv3, B, 32, 32);
    { int tot = B * 64 * 64 * 128;
      k_up2<<<launch1d(tot), TPB, 0, stream>>>(tA, tB, B, 32, 32, 128, tot);
      run_conv(stream, tB, tC, up3, B, 64, 64, 1);
      k_add<<<launch1d(tot), TPB, 0, stream>>>(tC, c3b, comb, tot); }
    // level 2 (64x64 -> 128x128, 128->64): final comb at 128^2 x 64
    basic3(stream, comb, tA, tB, tA, deconv2, B, 64, 64);
    { int tot = B * 128 * 128 * 64;
      k_up2<<<launch1d(tot), TPB, 0, stream>>>(tA, tB, B, 64, 64, 64, tot);
      run_conv(stream, tB, tC, up2w, B, 128, 128, 1);
      k_add<<<launch1d(tot), TPB, 0, stream>>>(tC, c2b, comb, tot); }

    // --- kernel heads + separable apply -----------------------------------
    float* out = (float*)d_out;
    const int septot = B * 3 * (int)HW0;

    subnet(stream, comb, tA, tB, Vb, v1, B, 128, 128);
    subnet(stream, comb, tA, tB, Hb, h1, B, 128, 128);
    k_sepconv<<<launch1d(septot), TPB, 0, stream>>>(
        frames, Vb, Hb, out, B, F, /*frame=*/0, H0, W0, /*accumulate=*/0, septot);

    subnet(stream, comb, tA, tB, Vb, v2, B, 128, 128);
    subnet(stream, comb, tA, tB, Hb, h2, B, 128, 128);
    k_sepconv<<<launch1d(septot), TPB, 0, stream>>>(
        frames, Vb, Hb, out, B, F, /*frame=*/1, H0, W0, /*accumulate=*/1, septot);
}
